// LigerFusedLinearDPOLoss_46720654246580
// MI455X (gfx1250) — matrix operand order repaired and checked by reading, not compiled, not run on hardware
//
#include <hip/hip_runtime.h>

#define H_DIM 2048
#define V_DIM 32000
#define NTOK 4096          // 8 * 512
#define NSEQ 8
#define SEQ_T 512
#define BLOCK_ROWS 128
#define NT 128             // column tile
#define KT 64              // K panel staged in LDS
#define NPANEL (H_DIM / KT)   // 32
#define VCH 10             // V chunks
#define CHUNK (V_DIM / VCH)   // 3200
#define TILES (CHUNK / NT)    // 25
#define LDS_STRIDE 88         // ushorts/row: 64 data + 24 pad (176B, 16B-aligned)
#define BETA_F 0.1f
#define IGNORE_INDEX (-100)

typedef __bf16 v16bf __attribute__((ext_vector_type(16)));
typedef float  v8f   __attribute__((ext_vector_type(8)));

union FragBF { v16bf v; uint4 q[2]; };
union AccF   { v8f v; float f[8]; };

#if __has_builtin(__builtin_amdgcn_sched_group_barrier)
#define SCHED_DS(n)   __builtin_amdgcn_sched_group_barrier(0x100, (n), 0)  // DS reads
#define SCHED_WMMA(n) __builtin_amdgcn_sched_group_barrier(0x008, (n), 0)  // MFMA/WMMA
#else
#define SCHED_DS(n)
#define SCHED_WMMA(n)
#endif

__device__ inline unsigned short f2bf(float x) {
  union { float f; unsigned int u; } c; c.f = x;
  unsigned int u = c.u;
  unsigned int r = u + 0x7FFFu + ((u >> 16) & 1u);   // round-to-nearest-even
  return (unsigned short)(r >> 16);
}

__device__ inline unsigned lds_off(const void* p) {
  // flat-aperture low 32 bits == wave-relative LDS byte address (ISA 10.2)
  return (unsigned)(unsigned long long)p;
}

// Issue one 64B-per-thread async panel slice for A and B (4+4 b128 moves).
// The instruction immediate offset applies to BOTH lds and global addresses.
__device__ inline void issue_async_panel(unsigned aA, unsigned aB,
                                         unsigned voffA, unsigned voffB,
                                         const unsigned short* Xq,
                                         const unsigned short* Wq) {
  asm volatile(
      "global_load_async_to_lds_b128 %0, %1, %2 offset:0\n\t"
      "global_load_async_to_lds_b128 %0, %1, %2 offset:16\n\t"
      "global_load_async_to_lds_b128 %0, %1, %2 offset:32\n\t"
      "global_load_async_to_lds_b128 %0, %1, %2 offset:48"
      :: "v"(aA), "v"(voffA), "s"(Xq) : "memory");
  asm volatile(
      "global_load_async_to_lds_b128 %0, %1, %2 offset:0\n\t"
      "global_load_async_to_lds_b128 %0, %1, %2 offset:16\n\t"
      "global_load_async_to_lds_b128 %0, %1, %2 offset:32\n\t"
      "global_load_async_to_lds_b128 %0, %1, %2 offset:48"
      :: "v"(aB), "v"(voffB), "s"(Wq) : "memory");
}

__device__ inline void wait_async0() {
  asm volatile("s_wait_asynccnt 0x0" ::: "memory");
}

// f32 -> packed bf16 conversion (one-shot pass over W and X)
__global__ __launch_bounds__(256) void cvt_f32_bf16(
    const float4* __restrict__ src, uint2* __restrict__ dst, int n4) {
  int i = blockIdx.x * 256 + threadIdx.x;
  if (i < n4) {
    float4 f = src[i];
    uint2 p;
    p.x = (unsigned)f2bf(f.x) | ((unsigned)f2bf(f.y) << 16);
    p.y = (unsigned)f2bf(f.z) | ((unsigned)f2bf(f.w) << 16);
    dst[i] = p;
  }
}

template <bool PRE>   // PRE: pre-converted bf16 + async double-buffered staging
__global__ __launch_bounds__(256) void dpo_gemm_partial(
    const float* __restrict__ Wf, const float* __restrict__ Xf,
    const unsigned short* __restrict__ Wq, const unsigned short* __restrict__ Xq,
    const int* __restrict__ target, const float* __restrict__ bias,
    float* __restrict__ m_part, float* __restrict__ s_part,
    float* __restrict__ t_part)
{
  __shared__ __align__(16) unsigned short lA[2][BLOCK_ROWS * LDS_STRIDE];
  __shared__ __align__(16) unsigned short lB[2][NT * LDS_STRIDE];

  const int rb = blockIdx.x;          // 0..31 row block
  const int ch = blockIdx.y;          // 0..9  V chunk
  const int rowbase = rb * BLOCK_ROWS;
  const int colchunk = ch * CHUNK;
  const int tid  = threadIdx.x;
  const int wave = tid >> 5;
  const int lane = tid & 31;
  const int half = lane >> 4;
  const int ln   = lane & 15;
  const int mrow0 = wave * 16;

  const int lr    = tid >> 1;         // loader: row 0..127
  const int lpart = tid & 1;          // loader: 64B half of a 128B row slice

  // LDS byte addresses of this thread's staging slice, per buffer
  unsigned aA[2], aB[2];
  aA[0] = lds_off(&lA[0][lr * LDS_STRIDE + lpart * 32]);
  aA[1] = lds_off(&lA[1][lr * LDS_STRIDE + lpart * 32]);
  aB[0] = lds_off(&lB[0][lr * LDS_STRIDE + lpart * 32]);
  aB[1] = lds_off(&lB[1][lr * LDS_STRIDE + lpart * 32]);
  const unsigned voffA0 = ((unsigned)(rowbase + lr) * H_DIM) * 2u + lpart * 64u;

  float rm[8], rs[8], tcap[8];
  int   tg[8];
#pragma unroll
  for (int i = 0; i < 8; ++i) {
    rm[i] = -INFINITY; rs[i] = 0.f; tcap[i] = 0.f;
    tg[i] = target[rowbase + mrow0 + i + half * 8];
  }

  for (int tile = 0; tile < TILES; ++tile) {
    const int colbase = colchunk + tile * NT;
    AccF acc[8];
#pragma unroll
    for (int nt = 0; nt < 8; ++nt)
#pragma unroll
      for (int i = 0; i < 8; ++i) acc[nt].f[i] = 0.f;

    // ---- compute for one staged panel (two 32-K WMMA steps) ----
    // Pipelined schedule per step: 6 ds_reads, 6x(1 WMMA + 2 ds_reads), 2 WMMA
    auto compute_panel = [&](int bi) {
      const unsigned short* pA = &lA[bi][0];
      const unsigned short* pB = &lB[bi][0];
#pragma unroll
      for (int s = 0; s < 2; ++s) {
        const int k0 = s * 32;
        FragBF a;   // A 16x32 bf16: M=ln, K = half*8 + {0..7, 16..23}
        {
          const int m  = mrow0 + ln;
          const int kb = k0 + half * 8;
          a.q[0] = *(const uint4*)(pA + m * LDS_STRIDE + kb);
          a.q[1] = *(const uint4*)(pA + m * LDS_STRIDE + kb + 16);
        }
        FragBF b[8];  // B 32x16 bf16: N=ln, K = half*8 + {0..7, 16..23}
#pragma unroll
        for (int nt = 0; nt < 8; ++nt) {
          const int n  = nt * 16 + ln;
          const int kb = k0 + half * 8;
          b[nt].q[0] = *(const uint4*)(pB + n * LDS_STRIDE + kb);
          b[nt].q[1] = *(const uint4*)(pB + n * LDS_STRIDE + kb + 16);
        }
#pragma unroll
        for (int nt = 0; nt < 8; ++nt)
          acc[nt].v = __builtin_amdgcn_wmma_f32_16x16x32_bf16(
              false, a.v, false, b[nt].v, (short)0, acc[nt].v, false, false);

        SCHED_DS(6);              // a.q[0..1], b0, b1 in flight first
#pragma unroll
        for (int g = 0; g < 6; ++g) {
          SCHED_WMMA(1);          // wmma nt=g ...
          SCHED_DS(2);            // ... overlapped with loads for b[g+2]
        }
        SCHED_WMMA(2);            // drain last two wmmas
      }
    };

    if (PRE) {
      // bf16 operands already in memory: async double-buffered staging
      const unsigned voffB0 = ((unsigned)(colbase + lr) * H_DIM) * 2u + lpart * 64u;
      issue_async_panel(aA[0], aB[0], voffA0, voffB0, Xq, Wq);
      wait_async0();
      __syncthreads();
#pragma unroll 1
      for (int pi = 0; pi < NPANEL; ++pi) {
        if (pi + 1 < NPANEL) {
          const unsigned kadd = (unsigned)(pi + 1) * (KT * 2u);
          issue_async_panel(aA[(pi + 1) & 1], aB[(pi + 1) & 1],
                            voffA0 + kadd, voffB0 + kadd, Xq, Wq);
        }
        compute_panel(pi & 1);        // overlaps with in-flight async loads
        wait_async0();
        __syncthreads();
      }
    } else {
      // fallback: stage + convert f32 -> bf16 synchronously (buffer 0 only)
      for (int kp = 0; kp < H_DIM; kp += KT) {
        __syncthreads();
        {
          const float* src = Xf + (size_t)(rowbase + lr) * H_DIM + kp + lpart * 32;
          unsigned short* dst = &lA[0][lr * LDS_STRIDE + lpart * 32];
#pragma unroll
          for (int j = 0; j < 8; ++j) {
            float4 f = *(const float4*)(src + j * 4);
            uint2 pk;
            pk.x = (unsigned)f2bf(f.x) | ((unsigned)f2bf(f.y) << 16);
            pk.y = (unsigned)f2bf(f.z) | ((unsigned)f2bf(f.w) << 16);
            *(uint2*)(dst + j * 4) = pk;
          }
        }
        {
          const float* src = Wf + (size_t)(colbase + lr) * H_DIM + kp + lpart * 32;
          if (kp + KT < H_DIM) __builtin_prefetch(src + KT, 0, 1);
          unsigned short* dst = &lB[0][lr * LDS_STRIDE + lpart * 32];
#pragma unroll
          for (int j = 0; j < 8; ++j) {
            float4 f = *(const float4*)(src + j * 4);
            uint2 pk;
            pk.x = (unsigned)f2bf(f.x) | ((unsigned)f2bf(f.y) << 16);
            pk.y = (unsigned)f2bf(f.z) | ((unsigned)f2bf(f.w) << 16);
            *(uint2*)(dst + j * 4) = pk;
          }
        }
        __syncthreads();
        compute_panel(0);
      }
    }

    // ---- tile epilogue: bias, target capture, online (max, sumexp) ----
    float tmax[8];
#pragma unroll
    for (int i = 0; i < 8; ++i) tmax[i] = -INFINITY;
#pragma unroll
    for (int nt = 0; nt < 8; ++nt) {
      const int col = colbase + nt * 16 + ln;
      const float bv = bias[col];
#pragma unroll
      for (int i = 0; i < 8; ++i) {
        float v = acc[nt].f[i] + bv;
        acc[nt].f[i] = v;
        tmax[i] = fmaxf(tmax[i], v);
        if (col == tg[i]) tcap[i] += v;
      }
    }
#pragma unroll
    for (int i = 0; i < 8; ++i) {
#pragma unroll
      for (int mk = 1; mk < 16; mk <<= 1)
        tmax[i] = fmaxf(tmax[i], __shfl_xor(tmax[i], mk, 32));
    }
#pragma unroll
    for (int i = 0; i < 8; ++i) {
      const float mn = fmaxf(rm[i], tmax[i]);
      float p = 0.f;
#pragma unroll
      for (int nt = 0; nt < 8; ++nt) p += __expf(acc[nt].f[i] - mn);
#pragma unroll
      for (int mk = 1; mk < 16; mk <<= 1) p += __shfl_xor(p, mk, 32);
      rs[i] = rs[i] * __expf(rm[i] - mn) + p;
      rm[i] = mn;
    }
  }

#pragma unroll
  for (int i = 0; i < 8; ++i) {
#pragma unroll
    for (int mk = 1; mk < 16; mk <<= 1) tcap[i] += __shfl_xor(tcap[i], mk, 32);
  }

  if (ln == 0) {   // lanes 0 and 16 write their half's 8 rows
#pragma unroll
    for (int i = 0; i < 8; ++i) {
      const int row = rowbase + mrow0 + i + half * 8;
      m_part[row * VCH + ch] = rm[i];
      s_part[row * VCH + ch] = rs[i];
      t_part[row * VCH + ch] = tcap[i];
    }
  }
}

__global__ __launch_bounds__(256) void dpo_finalize(
    const int* __restrict__ target,
    const float* __restrict__ m_part, const float* __restrict__ s_part,
    const float* __restrict__ t_part, float* __restrict__ out)
{
  __shared__ float ssum[NSEQ];
  __shared__ float scnt[NSEQ];
  const int tid = threadIdx.x;
  if (tid < NSEQ) { ssum[tid] = 0.f; scnt[tid] = 0.f; }
  __syncthreads();

  for (int row = tid; row < NTOK; row += 256) {
    float M = -INFINITY;
#pragma unroll
    for (int c = 0; c < VCH; ++c) M = fmaxf(M, m_part[row * VCH + c]);
    float S = 0.f, tgv = 0.f;
#pragma unroll
    for (int c = 0; c < VCH; ++c) {
      S   += s_part[row * VCH + c] * __expf(m_part[row * VCH + c] - M);
      tgv += t_part[row * VCH + c];
    }
    const float lse = M + __logf(S);
    const bool mask = (target[row] != IGNORE_INDEX);
    const float logp = mask ? (tgv - lse) : 0.f;
    const int seq = row / SEQ_T;
    atomicAdd(&ssum[seq], logp);
    if (mask) atomicAdd(&scnt[seq], 1.f);
  }
  __syncthreads();

  if (tid == 0) {
    float avg[NSEQ];
#pragma unroll
    for (int b = 0; b < NSEQ; ++b) avg[b] = ssum[b] / scnt[b];
    float align = 0.f, csum = 0.f, ccnt = 0.f;
#pragma unroll
    for (int b = 0; b < 4; ++b) {
      const float d = BETA_F * (avg[b] - avg[b + 4]);
      const float l = (d > 0.f) ? __logf(1.f + __expf(-d))
                                : (-d + __logf(1.f + __expf(d)));
      align += l;
      csum += ssum[b];
      ccnt += scnt[b];
    }
    align *= 0.25f;
    const float nll = -csum / ccnt;
    out[0] = align + nll;
  }
}

extern "C" void kernel_launch(void* const* d_in, const int* in_sizes, int n_in,
                              void* d_out, int out_size, void* d_ws, size_t ws_size,
                              hipStream_t stream) {
  const float* W      = (const float*)d_in[0];   // lin_weight (V, H)
  const float* X      = (const float*)d_in[1];   // _input    (8, 512, H)
  const int*   target = (const int*)d_in[2];     // (8, 512)
  const float* bias   = (const float*)d_in[3];   // (V,)

  float* ws = (float*)d_ws;
  float* m_part = ws;
  float* s_part = ws + (size_t)NTOK * VCH;
  float* t_part = ws + (size_t)2 * NTOK * VCH;   // partials: ~480 KB

  const size_t off_x = 512 * 1024;                          // 16B-aligned
  const size_t off_w = off_x + (size_t)NTOK * H_DIM * 2;    // X bf16: 16 MB
  const size_t need  = off_w + (size_t)V_DIM * H_DIM * 2;   // W bf16: 131 MB

  dim3 grid(NTOK / BLOCK_ROWS, VCH);             // 32 x 10 workgroups

  if (ws_size >= need) {
    unsigned short* Xq = (unsigned short*)((char*)d_ws + off_x);
    unsigned short* Wq = (unsigned short*)((char*)d_ws + off_w);
    const int n4x = NTOK * H_DIM / 4;
    const int n4w = V_DIM * H_DIM / 4;
    cvt_f32_bf16<<<(n4x + 255) / 256, 256, 0, stream>>>(
        (const float4*)X, (uint2*)Xq, n4x);
    cvt_f32_bf16<<<(n4w + 255) / 256, 256, 0, stream>>>(
        (const float4*)W, (uint2*)Wq, n4w);
    dpo_gemm_partial<true><<<grid, 256, 0, stream>>>(
        nullptr, nullptr, Wq, Xq, target, bias, m_part, s_part, t_part);
  } else {
    dpo_gemm_partial<false><<<grid, 256, 0, stream>>>(
        W, X, nullptr, nullptr, target, bias, m_part, s_part, t_part);
  }
  dpo_finalize<<<1, 256, 0, stream>>>(target, m_part, s_part, t_part,
                                      (float*)d_out);
}
